// LightAttention_18339510354098
// MI455X (gfx1250) — compile-verified, matrix operand order checked by reading, not run on hardware
//
#include <hip/hip_runtime.h>

// ---------------------------------------------------------------------------
// LightAttention for MI455X (gfx1250, wave32, WMMA f32_16x16x32_f16 + TDM)
// ---------------------------------------------------------------------------
constexpr int B = 8, T = 64, P = 196, M = 2048, H = 768;
constexpr int PPAD = 224;                 // P rounded up to multiple of 32 (K tile)
constexpr int KC   = 1024;                // K chunk staged in LDS via TDM (32 KB)

typedef __attribute__((ext_vector_type(16))) _Float16 v16h;
typedef __attribute__((ext_vector_type(8)))  _Float16 v8h;
typedef __attribute__((ext_vector_type(8)))  float    v8f;
typedef unsigned int u32;
typedef __attribute__((ext_vector_type(4))) u32 u32x4;
typedef __attribute__((ext_vector_type(8))) int i32x8;
typedef __attribute__((ext_vector_type(4))) int i32x4;

// Workspace layout (bytes, all 16B aligned)
constexpr size_t OFF_FMA   = 0;                                   // fp16 [B][P][M]
constexpr size_t OFF_FMT   = OFF_FMA + (size_t)B * P * M * 2;     // fp16 [B][M][PPAD]
constexpr size_t OFF_WMAP  = OFF_FMT + (size_t)B * M * PPAD * 2;  // fp16 [H][M]
constexpr size_t OFF_WFIN  = OFF_WMAP + (size_t)H * M * 2;        // fp16 [H][M]
constexpr size_t OFF_MLO   = OFF_WFIN + (size_t)H * M * 2;        // fp32 [B][P][H]
constexpr size_t OFF_ATT16 = OFF_MLO + (size_t)B * P * H * 4;     // fp16 [B][T][PPAD]
constexpr size_t OFF_CTX16 = OFF_ATT16 + (size_t)B * T * PPAD * 2;// fp16 [B][T][M]

// ---------------------------------------------------------------------------
// TDM: 2D tile (fp16) global -> LDS, per CDNA5 D# layout (ISA 08 §8.3/8.4).
//   tile_d0 elements per row, tile_d1 rows; rows beyond tensor_d1 read as 0.
// ---------------------------------------------------------------------------
__device__ __forceinline__ void tdm_load_2d_f16(const void* gaddr, u32 lds_off,
                                                u32 tile_d0, u32 tile_d1,
                                                u32 tensor_d0, u32 tensor_d1,
                                                u32 row_stride) {
  unsigned long long ga = (unsigned long long)gaddr;
  u32x4 g0;
  g0.x = 1u;                                  // count=1, user load descriptor
  g0.y = lds_off;                             // lds_addr (bytes)
  g0.z = (u32)ga;                             // global_addr[31:0]
  g0.w = (u32)(ga >> 32) | (2u << 30);        // global_addr[56:32] | type=2
  i32x8 g1;
  g1[0] = (int)(1u << 16);                    // workgroup_mask=0, data_size=1 (2B)
  g1[1] = (int)(tensor_d0 << 16);             // tensor_dim0[15:0] -> [31:16]
  g1[2] = (int)((tensor_d0 >> 16) | (tensor_d1 << 16));
  g1[3] = (int)((tensor_d1 >> 16) | (tile_d0 << 16));
  g1[4] = (int)tile_d1;                       // tile_dim1 (tile_dim2=0)
  g1[5] = (int)row_stride;                    // tensor_dim0_stride[31:0]
  g1[6] = 0;                                  // stride hi, dim1_stride lo
  g1[7] = 0;
  i32x4 z4 = {0, 0, 0, 0};
#if defined(__clang_major__) && __clang_major__ >= 23
  i32x8 z8 = {0, 0, 0, 0, 0, 0, 0, 0};
  __builtin_amdgcn_tensor_load_to_lds(g0, g1, z4, z4, z8, 0);
#else
  __builtin_amdgcn_tensor_load_to_lds(g0, g1, z4, z4, 0);
#endif
}

// Load one 16x32 fp16 WMMA operand fragment (row-major, K contiguous).
// lane L holds row (caller passes absolute row = tilebase + L%16);
// K offsets = k0 + (L/16)*8 + {0..7} and +16 more.
__device__ __forceinline__ v16h load_frag(const _Float16* mat, int ld, int row,
                                          int k0, int lane) {
  const _Float16* p = mat + (size_t)row * ld + k0 + ((lane >> 4) << 3);
  v8h lo = *(const v8h*)(p);
  v8h hi = *(const v8h*)(p + 16);
  return __builtin_shufflevector(lo, hi, 0, 1, 2, 3, 4, 5, 6, 7,
                                          8, 9, 10, 11, 12, 13, 14, 15);
}

// ---------------------------------------------------------------------------
// Stage 0: fp32 -> fp16 converts (and padded transpose of fm)
// ---------------------------------------------------------------------------
__global__ void k_cvt(const float* __restrict__ src, _Float16* __restrict__ dst, int n) {
  int i = blockIdx.x * blockDim.x + threadIdx.x;
  if (i < n) dst[i] = (_Float16)src[i];
}

__global__ void k_fmT(const float* __restrict__ maps, _Float16* __restrict__ fmT) {
  int i = blockIdx.x * blockDim.x + threadIdx.x;
  if (i >= B * M * PPAD) return;
  int p = i % PPAD;
  int m = (i / PPAD) % M;
  int b = i / (PPAD * M);
  float v = (p < P) ? maps[((size_t)b * P + p) * M + m] : 0.0f;
  fmT[i] = (_Float16)v;
}

// ---------------------------------------------------------------------------
// Stage 1: mlo[b][p][h] = fm[b,p,:] . W_map[h,:] + b_map[h]   (K = 2048)
// Block = 4 waves. A tile (16 x KC) staged in LDS by TDM, shared by all waves.
// Each wave computes a 16x64 C tile (4 accumulators, A-fragment reuse).
// ---------------------------------------------------------------------------
__global__ void __launch_bounds__(128)
k_mlo(const _Float16* __restrict__ fmA, const _Float16* __restrict__ Wm,
      const float* __restrict__ b_map, float* __restrict__ mlo) {
  __shared__ _Float16 Atile[16 * KC];              // 32 KB
  int lane = threadIdx.x & 31;
  int wave = threadIdx.x >> 5;
  int b  = blockIdx.z;                             // 8
  int pt = blockIdx.y;                             // 13 p-tiles (208 >= 196)
  int hg = blockIdx.x * 4 + wave;                  // 12 groups of 64 h
  const _Float16* Abase = fmA + ((size_t)b * P + pt * 16) * M;
  int rows = P - pt * 16; if (rows > 16) rows = 16; // TDM zero-fills OOB rows

  int arow = lane & 15;
  int br0  = hg * 64 + (lane & 15);
  v8f c[4] = {};

  for (int chunk = 0; chunk < M / KC; ++chunk) {
    if (threadIdx.x < 32) {                        // wave 0 drives the DMA
      tdm_load_2d_f16(Abase + chunk * KC, (u32)(size_t)Atile,
                      KC, 16, KC, (u32)rows, M);
      __builtin_amdgcn_s_wait_tensorcnt((short)0);
    }
    __syncthreads();
    for (int k = 0; k < KC; k += 32) {
      int kg = chunk * KC + k;
      __builtin_prefetch(Wm + (size_t)br0 * M + kg + 128, 0, 3);
      v16h a = load_frag(Atile, KC, arow, k, lane); // ds_load_b128 x2
      v16h bb[4];
#pragma unroll
      for (int j = 0; j < 4; ++j)
        bb[j] = load_frag(Wm, M, br0 + j * 16, kg, lane);
#pragma unroll
      for (int j = 0; j < 4; ++j)
        c[j] = __builtin_amdgcn_wmma_f32_16x16x32_f16(false, a, false, bb[j],
                                                      (short)0, c[j], false, false);
    }
    __syncthreads();                               // all reads done before next DMA
  }

  int po = (lane >> 4) * 8;
#pragma unroll
  for (int j = 0; j < 4; ++j) {
    int h = hg * 64 + j * 16 + (lane & 15);
    float bias = b_map[h];
#pragma unroll
    for (int r = 0; r < 8; ++r) {
      int p = pt * 16 + po + r;
      if (p < P) mlo[((size_t)b * P + p) * H + h] = c[j][r] + bias;
    }
  }
}

// ---------------------------------------------------------------------------
// Stage 2: scores + softmax.  One block per (b,t).
// score[p] = sum_h W_rect[h]*relu(mlo[b,p,h]+hid[b,t,h]) + b_rect
// ReLU between add and reduction => VALU (cannot be a GEMM); ~0.3 Gops total.
// ---------------------------------------------------------------------------
__global__ void __launch_bounds__(256)
k_scores(const float* __restrict__ mlo, const float* __restrict__ hiddens,
         const float* __restrict__ W_rect, const float* __restrict__ b_rect,
         float* __restrict__ attn_out, _Float16* __restrict__ attn16) {
  __shared__ float hid[H];
  __shared__ float wr[H];
  __shared__ float red[256];
  int t = blockIdx.x, b = blockIdx.y;
  int tid = threadIdx.x;

  for (int h = tid; h < H; h += 256) {
    hid[h] = hiddens[((size_t)b * T + t) * H + h];
    wr[h]  = W_rect[h];
  }
  __syncthreads();

  float val = 0.0f;
  if (tid < P) {
    const float* mrow = mlo + ((size_t)b * P + tid) * H;
    float acc = 0.0f;
    for (int h = 0; h < H; ++h) {
      float r = mrow[h] + hid[h];
      acc = fmaf(wr[h], (r > 0.0f ? r : 0.0f), acc);
    }
    val = acc + b_rect[0];
  }

  red[tid] = (tid < P) ? val : -3.0e38f;
  __syncthreads();
  for (int s = 128; s > 0; s >>= 1) {
    if (tid < s) red[tid] = fmaxf(red[tid], red[tid + s]);
    __syncthreads();
  }
  float mx = red[0];
  __syncthreads();

  float e = (tid < P) ? __expf(val - mx) : 0.0f;
  red[tid] = e;
  __syncthreads();
  for (int s = 128; s > 0; s >>= 1) {
    if (tid < s) red[tid] += red[tid + s];
    __syncthreads();
  }
  float a = e / red[0];

  if (tid < P)    attn_out[((size_t)b * T + t) * P + tid] = a;
  if (tid < PPAD) attn16[((size_t)b * T + t) * PPAD + tid] =
                      (_Float16)((tid < P) ? a : 0.0f);
}

// ---------------------------------------------------------------------------
// Stage 3: ctx[b][t][m] = sum_p attn[b,t,p] * fm[b,p,m]   (K = PPAD = 224)
// A = attn16 t-tile staged in LDS via TDM (7 KB); B = fmT rows (m).
// Each wave computes 16t x 64m.
// ---------------------------------------------------------------------------
__global__ void __launch_bounds__(128)
k_ctx(const _Float16* __restrict__ attn16, const _Float16* __restrict__ fmT,
      _Float16* __restrict__ ctx16) {
  __shared__ _Float16 Atile[16 * PPAD];            // 7 KB
  int lane = threadIdx.x & 31;
  int wave = threadIdx.x >> 5;
  int b  = blockIdx.z;                             // 8
  int tt = blockIdx.y;                             // 4 t-tiles
  int mg = blockIdx.x * 4 + wave;                  // 32 groups of 64 m
  const _Float16* Abase = attn16 + ((size_t)b * T + tt * 16) * PPAD;
  const _Float16* Bf    = fmT    + (size_t)b * M * PPAD;

  if (threadIdx.x < 32) {
    tdm_load_2d_f16(Abase, (u32)(size_t)Atile, PPAD, 16, PPAD, 16, PPAD);
    __builtin_amdgcn_s_wait_tensorcnt((short)0);
  }
  __syncthreads();

  int arow = lane & 15;
  int br0  = mg * 64 + (lane & 15);
  v8f c[4] = {};
  for (int k = 0; k < PPAD; k += 32) {
    v16h a = load_frag(Atile, PPAD, arow, k, lane);
    v16h bb[4];
#pragma unroll
    for (int j = 0; j < 4; ++j)
      bb[j] = load_frag(Bf, PPAD, br0 + j * 16, k, lane);
#pragma unroll
    for (int j = 0; j < 4; ++j)
      c[j] = __builtin_amdgcn_wmma_f32_16x16x32_f16(false, a, false, bb[j],
                                                    (short)0, c[j], false, false);
  }

  int to = (lane >> 4) * 8;
#pragma unroll
  for (int j = 0; j < 4; ++j) {
    int m = mg * 64 + j * 16 + (lane & 15);
#pragma unroll
    for (int r = 0; r < 8; ++r) {
      int t = tt * 16 + to + r;
      ctx16[((size_t)b * T + t) * M + m] = (_Float16)c[j][r];
    }
  }
}

// ---------------------------------------------------------------------------
// Stage 4: out[b][t][h] = ctx[b,t,:] . W_final[h,:] + b_final[h] + hiddens[b,t,h]
// A = ctx16 t-tile staged in LDS via TDM; each wave computes 16t x 64h.
// ---------------------------------------------------------------------------
__global__ void __launch_bounds__(128)
k_final(const _Float16* __restrict__ ctx16, const _Float16* __restrict__ Wf,
        const float* __restrict__ b_final, const float* __restrict__ hiddens,
        float* __restrict__ out) {
  __shared__ _Float16 Atile[16 * KC];              // 32 KB
  int lane = threadIdx.x & 31;
  int wave = threadIdx.x >> 5;
  int b  = blockIdx.z;                             // 8
  int tt = blockIdx.y;                             // 4 t-tiles
  int hg = blockIdx.x * 4 + wave;                  // 12 groups of 64 h
  const _Float16* Abase = ctx16 + ((size_t)b * T + tt * 16) * M;

  int arow = lane & 15;
  int br0  = hg * 64 + (lane & 15);
  v8f c[4] = {};

  for (int chunk = 0; chunk < M / KC; ++chunk) {
    if (threadIdx.x < 32) {
      tdm_load_2d_f16(Abase + chunk * KC, (u32)(size_t)Atile,
                      KC, 16, KC, 16, M);
      __builtin_amdgcn_s_wait_tensorcnt((short)0);
    }
    __syncthreads();
    for (int k = 0; k < KC; k += 32) {
      int kg = chunk * KC + k;
      __builtin_prefetch(Wf + (size_t)br0 * M + kg + 128, 0, 3);
      v16h a = load_frag(Atile, KC, arow, k, lane);
      v16h bb[4];
#pragma unroll
      for (int j = 0; j < 4; ++j)
        bb[j] = load_frag(Wf, M, br0 + j * 16, kg, lane);
#pragma unroll
      for (int j = 0; j < 4; ++j)
        c[j] = __builtin_amdgcn_wmma_f32_16x16x32_f16(false, a, false, bb[j],
                                                      (short)0, c[j], false, false);
    }
    __syncthreads();
  }

  int to = (lane >> 4) * 8;
#pragma unroll
  for (int j = 0; j < 4; ++j) {
    int h = hg * 64 + j * 16 + (lane & 15);
    float bias = b_final[h];
#pragma unroll
    for (int r = 0; r < 8; ++r) {
      int t = tt * 16 + to + r;
      size_t idx = ((size_t)b * T + t) * H + h;
      out[idx] = c[j][r] + bias + hiddens[idx];
    }
  }
}

// ---------------------------------------------------------------------------
extern "C" void kernel_launch(void* const* d_in, const int* in_sizes, int n_in,
                              void* d_out, int out_size, void* d_ws, size_t ws_size,
                              hipStream_t stream) {
  (void)in_sizes; (void)n_in; (void)out_size; (void)ws_size;
  const float* maps    = (const float*)d_in[0];
  const float* hiddens = (const float*)d_in[1];
  const float* W_map   = (const float*)d_in[2];
  const float* b_map   = (const float*)d_in[3];
  const float* W_final = (const float*)d_in[4];
  const float* b_final = (const float*)d_in[5];
  const float* W_rect  = (const float*)d_in[6];
  const float* b_rect  = (const float*)d_in[7];

  float* out  = (float*)d_out;                     // (B,T,H)
  float* attn = out + (size_t)B * T * H;           // (B,T,P)

  char* ws = (char*)d_ws;
  _Float16* fmA   = (_Float16*)(ws + OFF_FMA);
  _Float16* fmT   = (_Float16*)(ws + OFF_FMT);
  _Float16* Wm16  = (_Float16*)(ws + OFF_WMAP);
  _Float16* Wf16  = (_Float16*)(ws + OFF_WFIN);
  float*    mlo   = (float*)   (ws + OFF_MLO);
  _Float16* att16 = (_Float16*)(ws + OFF_ATT16);
  _Float16* ctx16 = (_Float16*)(ws + OFF_CTX16);

  // Stage 0: converts
  {
    int n = B * P * M;
    k_cvt<<<(n + 255) / 256, 256, 0, stream>>>(maps, fmA, n);
  }
  {
    int n = H * M;
    k_cvt<<<(n + 255) / 256, 256, 0, stream>>>(W_map, Wm16, n);
    k_cvt<<<(n + 255) / 256, 256, 0, stream>>>(W_final, Wf16, n);
  }
  {
    int n = B * M * PPAD;
    k_fmT<<<(n + 255) / 256, 256, 0, stream>>>(maps, fmT);
  }

  // Stage 1: map_linear GEMM (TDM + WMMA)
  k_mlo<<<dim3(3, 13, 8), 128, 0, stream>>>(fmA, Wm16, b_map, mlo);

  // Stage 2: fused rectified score + softmax (VALU)
  k_scores<<<dim3(T, B), 256, 0, stream>>>(mlo, hiddens, W_rect, b_rect, attn, att16);

  // Stage 3: attention-weighted context GEMM (TDM + WMMA)
  k_ctx<<<dim3(8, 4, 8), 128, 0, stream>>>(att16, fmT, ctx16);

  // Stage 4: output GEMM + bias + residual (TDM + WMMA)
  k_final<<<dim3(3, 4, 8), 128, 0, stream>>>(ctx16, Wf16, b_final, hiddens, out);
}
